// WindowAttention_24438363914431
// MI455X (gfx1250) — matrix operand order repaired and checked by reading, not compiled, not run on hardware
//
#include <hip/hip_runtime.h>
#include <hip/hip_bf16.h>

typedef __attribute__((ext_vector_type(16))) _Float16 v16h;
typedef __attribute__((ext_vector_type(8)))  _Float16 v8h;
typedef __attribute__((ext_vector_type(8)))  float    v8f;

#define WMMA_F16(a, b, c) \
  __builtin_amdgcn_wmma_f32_16x16x32_f16(false, (a), false, (b), (short)0, (c), false, false)

// ---- fragment loaders following CDNA5 ISA VGPR layouts (CDNA5 ISA §7.12.2) ----
// A (16x32 f16, MxK): lanes 0-15 row M=lane&15, K = {kh*8..kh*8+7, 16+kh*8..+7}, kh=lane>=16
static __device__ inline v16h frag_a_f16(const _Float16* base, int ld, int lane) {
  const int m = lane & 15, kh = (lane >> 4) & 1;
  const _Float16* r = base + (size_t)m * ld + kh * 8;
  union { v16h v; v8h h[2]; } u;
  u.h[0] = *(const v8h*)(r);
  u.h[1] = *(const v8h*)(r + 16);
  return u.v;
}
// B (32x16 f16, KxN), given B^T storage (per-column contiguous K): col N=lane&15, K = {kh*8..kh*8+7, 16+kh*8..+7}
static __device__ inline v16h frag_b_f16(const _Float16* baseT, int ld, int lane) {
  const int n = lane & 15, kh = (lane >> 4) & 1;
  const _Float16* r = baseT + (size_t)n * ld + kh * 8;
  union { v16h v; v8h h[2]; } u;
  u.h[0] = *(const v8h*)(r);
  u.h[1] = *(const v8h*)(r + 16);
  return u.v;
}

// ================= Kernel 0: one-shot f32 -> f16 weight conversion =================
__global__ __launch_bounds__(256) void cvt_weights(
    const float* __restrict__ qkv_w, const float* __restrict__ proj_w,
    _Float16* __restrict__ Wq, _Float16* __restrict__ Wp) {
  const int i = blockIdx.x * 256 + threadIdx.x;
  if (i < 576 * 192) Wq[i] = (_Float16)qkv_w[i];
  if (i < 192 * 192) Wp[i] = (_Float16)proj_w[i];
}

// ================= Kernel 1: QKV GEMM (+bias, q pre-scaled, V transposed) =================
// x:[262144,192] f32 staged once per block as f16 in LDS; Wq:[576,192] f16 (== B^T).
// Block = 128 thr (4 waves) owns 64 rows, loops over 9 column groups. Grid = 4096.
__global__ __launch_bounds__(128) void qkv_kernel(
    const float* __restrict__ x, const _Float16* __restrict__ Wq,
    const float* __restrict__ qkv_b,
    _Float16* __restrict__ Qh, _Float16* __restrict__ Kh, _Float16* __restrict__ Vth) {
  __shared__ __align__(16) _Float16 Xs[64 * 192];
  const int tid = threadIdx.x;
  const int lane = tid & 31, wave = tid >> 5;
  const int m0 = blockIdx.x * 64;

  // stage + convert the contiguous 64x192 f32 x-tile -> f16 LDS (read x exactly once)
  {
    const float* xb = x + (size_t)m0 * 192;
    for (int i = tid * 8; i < 64 * 192; i += 128 * 8) {
      v8h o;
#pragma unroll
      for (int j = 0; j < 8; ++j) o[j] = (_Float16)xb[i + j];
      *(v8h*)(Xs + i) = o;
    }
  }
  __syncthreads();

  const float scale = 0.17677669529663687f;  // 32^-0.5
  const int hi = (lane >> 4) & 1;
  const _Float16* As = Xs + wave * 16 * 192;

  for (int g = 0; g < 9; ++g) {
    const int n0 = g * 64;
    v8f acc[4] = {};
#pragma unroll
    for (int kt = 0; kt < 6; ++kt) {
      v16h a = frag_a_f16(As + kt * 32, 192, lane);
#pragma unroll
      for (int ct = 0; ct < 4; ++ct) {
        v16h bb = frag_b_f16(Wq + (size_t)(n0 + ct * 16) * 192 + kt * 32, 192, lane);
        acc[ct] = WMMA_F16(a, bb, acc[ct]);
      }
    }
#pragma unroll
    for (int ct = 0; ct < 4; ++ct) {
      const int n = n0 + ct * 16 + (lane & 15);
      const int p = n / 192, rem = n % 192;
      const int h = rem >> 5, d = rem & 31;
      const float bias = qkv_b[n];
#pragma unroll
      for (int r = 0; r < 8; ++r) {
        const int m = m0 + wave * 16 + r + hi * 8;
        const int b_ = m >> 6, tok = m & 63;
        const size_t bh = (size_t)b_ * 6 + h;
        const float v = acc[ct][r] + bias;
        if (p == 0)      Qh[(bh * 64 + tok) * 32 + d] = (_Float16)(v * scale);
        else if (p == 1) Kh[(bh * 64 + tok) * 32 + d] = (_Float16)v;
        else             Vth[(bh * 32 + d) * 64 + tok] = (_Float16)v;
      }
    }
  }
}

// ================= Kernel 2: fused attention per (window, head) =================
// grid (6, 4096), block 128 (4 waves). S=Q*K^T (+rel-pos bias), softmax, O=P*V.
__global__ __launch_bounds__(128) void attn_kernel(
    const _Float16* __restrict__ Qh, const _Float16* __restrict__ Kh,
    const _Float16* __restrict__ Vth, const float* __restrict__ rpb_table,
    _Float16* __restrict__ Og) {
  __shared__ __align__(16) _Float16 Qs[64 * 32];
  __shared__ __align__(16) _Float16 Ks[64 * 32];
  __shared__ __align__(16) _Float16 Vts[32 * 64];
  __shared__ float Sf[64 * 65];                 // pad 65 -> conflict-free softmax rows
  __shared__ __align__(16) _Float16 Pf[64 * 64];
  __shared__ float rpbs[225 * 6];

  const int tid = threadIdx.x;
  const int lane = tid & 31, wave = tid >> 5;
  const int h = blockIdx.x;
  const int b = blockIdx.y;
  const size_t bh = (size_t)b * 6 + h;

  // stage Q/K/V^T tiles (2048 halfs each = 256 uint4) + bias table
  {
    const uint4* sq = (const uint4*)(Qh + bh * 2048);
    const uint4* sk = (const uint4*)(Kh + bh * 2048);
    const uint4* sv = (const uint4*)(Vth + bh * 2048);
    uint4* dq = (uint4*)Qs; uint4* dk = (uint4*)Ks; uint4* dv = (uint4*)Vts;
    for (int i = tid; i < 256; i += 128) { dq[i] = sq[i]; dk[i] = sk[i]; dv[i] = sv[i]; }
    for (int i = tid; i < 225 * 6; i += 128) rpbs[i] = rpb_table[i];
  }
  __syncthreads();

  // ---- S = Q_scaled @ K^T : wave owns 16 rows, 4 col-tiles, K=32 => 1 WMMA each ----
  v16h a = frag_a_f16(Qs + wave * 16 * 32, 32, lane);
  v8f s[4];
#pragma unroll
  for (int ct = 0; ct < 4; ++ct) {
    v16h kb = frag_b_f16(Ks + ct * 16 * 32, 32, lane);
    v8f z = {};
    s[ct] = WMMA_F16(a, kb, z);
  }
  const int hi = (lane >> 4) & 1;
#pragma unroll
  for (int ct = 0; ct < 4; ++ct) {
    const int col = ct * 16 + (lane & 15);
    const int mh_ = col >> 3, mw_ = col & 7;
#pragma unroll
    for (int r = 0; r < 8; ++r) {
      const int row = wave * 16 + r + hi * 8;
      const int nh_ = row >> 3, nw_ = row & 7;
      const int idx = (nh_ - mh_ + 7) * 15 + (nw_ - mw_ + 7);
      Sf[row * 65 + col] = s[ct][r] + rpbs[idx * 6 + h];
    }
  }
  __syncthreads();

  // ---- softmax per row, re-quantize to f16 P ----
  if (tid < 64) {
    const int base = tid * 65;
    float mx = -3.4e38f;
    for (int i = 0; i < 64; ++i) mx = fmaxf(mx, Sf[base + i]);
    float sum = 0.f;
    for (int i = 0; i < 64; ++i) { float e = __expf(Sf[base + i] - mx); Sf[base + i] = e; sum += e; }
    const float rs = 1.f / sum;
    for (int i = 0; i < 64; ++i) Pf[tid * 64 + i] = (_Float16)(Sf[base + i] * rs);
  }
  __syncthreads();

  // ---- O = P @ V : wave owns 16 rows; 2 d-tiles x 2 K-steps ----
#pragma unroll
  for (int ct = 0; ct < 2; ++ct) {
    v8f acc = {};
#pragma unroll
    for (int kk = 0; kk < 2; ++kk) {
      v16h pa = frag_a_f16(Pf + wave * 16 * 64 + kk * 32, 64, lane);
      v16h vb = frag_b_f16(Vts + ct * 16 * 64 + kk * 32, 64, lane);
      acc = WMMA_F16(pa, vb, acc);
    }
    const int d = ct * 16 + (lane & 15);
#pragma unroll
    for (int r = 0; r < 8; ++r) {
      const int tok = wave * 16 + r + hi * 8;
      Og[((size_t)b * 64 + tok) * 192 + h * 32 + d] = (_Float16)acc[r];
    }
  }
}

// ================= Kernel 3: output projection =================
// Og:[262144,192] f16 staged once per block in LDS; Wp:[192,192] f16 (== B^T). Grid = 4096.
__global__ __launch_bounds__(128) void proj_kernel(
    const _Float16* __restrict__ Og, const _Float16* __restrict__ Wp,
    const float* __restrict__ proj_b, float* __restrict__ out) {
  __shared__ __align__(16) _Float16 Os[64 * 192];
  const int tid = threadIdx.x;
  const int lane = tid & 31, wave = tid >> 5;
  const int m0 = blockIdx.x * 64;

  // stage the contiguous 64x192 f16 O-tile (1536 uint4)
  {
    const uint4* src = (const uint4*)(Og + (size_t)m0 * 192);
    uint4* dst = (uint4*)Os;
    for (int i = tid; i < 1536; i += 128) dst[i] = src[i];
  }
  __syncthreads();

  const int hi = (lane >> 4) & 1;
  const _Float16* As = Os + wave * 16 * 192;

  for (int g = 0; g < 3; ++g) {
    const int n0 = g * 64;
    v8f acc[4] = {};
#pragma unroll
    for (int kt = 0; kt < 6; ++kt) {
      v16h a = frag_a_f16(As + kt * 32, 192, lane);
#pragma unroll
      for (int ct = 0; ct < 4; ++ct) {
        v16h bb = frag_b_f16(Wp + (size_t)(n0 + ct * 16) * 192 + kt * 32, 192, lane);
        acc[ct] = WMMA_F16(a, bb, acc[ct]);
      }
    }
#pragma unroll
    for (int ct = 0; ct < 4; ++ct) {
      const int n = n0 + ct * 16 + (lane & 15);
      const float bias = proj_b[n];
#pragma unroll
      for (int r = 0; r < 8; ++r) {
        const int m = m0 + wave * 16 + r + hi * 8;
        out[(size_t)m * 192 + n] = acc[ct][r] + bias;
      }
    }
  }
}

extern "C" void kernel_launch(void* const* d_in, const int* in_sizes, int n_in,
                              void* d_out, int out_size, void* d_ws, size_t ws_size,
                              hipStream_t stream) {
  const float* x      = (const float*)d_in[0];
  const float* qkv_w  = (const float*)d_in[1];
  const float* qkv_b  = (const float*)d_in[2];
  const float* proj_w = (const float*)d_in[3];
  const float* proj_b = (const float*)d_in[4];
  const float* rpb    = (const float*)d_in[5];

  char* ws = (char*)d_ws;
  const size_t SZ = (size_t)4096 * 6 * 64 * 32 * sizeof(_Float16);  // 100,663,296 B
  _Float16* Qh  = (_Float16*)(ws);
  _Float16* Kh  = (_Float16*)(ws + SZ);
  _Float16* Vth = (_Float16*)(ws + 2 * SZ);
  _Float16* Og  = (_Float16*)(ws + 3 * SZ);
  _Float16* Wq  = (_Float16*)(ws + 4 * SZ);                         // 576*192 f16
  _Float16* Wp  = (_Float16*)(ws + 4 * SZ + (size_t)576 * 192 * 2); // 192*192 f16

  cvt_weights<<<dim3((576 * 192 + 255) / 256), 256, 0, stream>>>(qkv_w, proj_w, Wq, Wp);
  qkv_kernel<<<dim3(4096), 128, 0, stream>>>(x, Wq, qkv_b, Qh, Kh, Vth);
  attn_kernel<<<dim3(6, 4096), 128, 0, stream>>>(Qh, Kh, Vth, rpb, Og);
  proj_kernel<<<dim3(4096), 128, 0, stream>>>(Og, Wp, proj_b, (float*)d_out);
}
